// Salient2BroadModule_43997644980503
// MI455X (gfx1250) — matrix operand order repaired and checked by reading, not compiled, not run on hardware
//
#include <hip/hip_runtime.h>

typedef __attribute__((ext_vector_type(16))) _Float16 v16h;
typedef __attribute__((ext_vector_type(8)))  _Float16 v8h;
typedef __attribute__((ext_vector_type(8)))  float    v8f;

// ---------------------------------------------------------------------------
// Problem constants
// ---------------------------------------------------------------------------
#define B_   64
#define C_   1024
#define T_   4
#define H_   32
#define W_   16
#define HW_  512        // H_*W_
#define TL_  3          // t - (POS+1)
#define KK_  9          // K*K
#define ALPHA_ 0.1f
#define BETA_  5.0f

// ---------------------------------------------------------------------------
// Async global->LDS (CDNA5). Guarded; falls back to sync copy.
// ---------------------------------------------------------------------------
#if defined(__has_builtin)
#if __has_builtin(__builtin_amdgcn_global_load_async_to_lds_b32) && \
    __has_builtin(__builtin_amdgcn_s_wait_asynccnt)
#define USE_ASYNC_LDS 1
#endif
#endif

__device__ __forceinline__ void issueTileB32(float* ldsDst, const float* gsrc) {
#ifdef USE_ASYNC_LDS
    __builtin_amdgcn_global_load_async_to_lds_b32(
        (__attribute__((address_space(1))) int*)(gsrc),
        (__attribute__((address_space(3))) int*)(ldsDst), 0, 0);
#else
    *ldsDst = *gsrc;
#endif
}
__device__ __forceinline__ void asyncWait1() {
#ifdef USE_ASYNC_LDS
    __builtin_amdgcn_s_wait_asynccnt(1);
#endif
}
__device__ __forceinline__ void asyncWait0() {
#ifdef USE_ASYNC_LDS
    __builtin_amdgcn_s_wait_asynccnt(0);
#endif
}

// ---------------------------------------------------------------------------
// Shared-memory block reduction (sum)
// ---------------------------------------------------------------------------
__device__ __forceinline__ float blockSum(float v, float* red, int tid, int n) {
    red[tid] = v; __syncthreads();
    for (int s = n >> 1; s > 0; s >>= 1) {
        if (tid < s) red[tid] += red[tid + s];
        __syncthreads();
    }
    float r = red[0]; __syncthreads();
    return r;
}

// ---------------------------------------------------------------------------
// Pass 1: per-(b,c) sums: total (gap/cw), interior (cwo), t=0 slice (xf)
// grid = 65536 blocks, 256 threads; vectorized float4 loads
// ---------------------------------------------------------------------------
__global__ void k_pass1_reduce(const float* __restrict__ x,
                               float* __restrict__ gap_sum,
                               float* __restrict__ cwo_sum,
                               float* __restrict__ xf_sum) {
    int bc = blockIdx.x;
    int tid = threadIdx.x;
    const float4* p4 = (const float4*)(x + (size_t)bc * (T_ * HW_));
    float tot = 0.f, inter = 0.f, xf = 0.f;
#pragma unroll
    for (int j = 0; j < 2; ++j) {
        int i4 = tid + j * 256;          // 0..511
        float4 v = p4[i4];
        int i = i4 * 4;
        int t = i >> 9;
        int n = i & (HW_ - 1);
        int y = n >> 4, xb = n & 15;     // 4 consecutive share t and y
        float vs[4] = {v.x, v.y, v.z, v.w};
#pragma unroll
        for (int e = 0; e < 4; ++e) {
            float vv = vs[e];
            tot += vv;
            if (t == 0) xf += vv;
            int xc = xb + e;
            if (y >= 1 && y <= H_ - 2 && xc >= 1 && xc <= W_ - 2) inter += vv;
        }
    }
    __shared__ float red[256];
    float Tt = blockSum(tot,   red, tid, 256);
    float Ii = blockSum(inter, red, tid, 256);
    float Ff = blockSum(xf,    red, tid, 256);
    if (tid == 0) { gap_sum[bc] = Tt; cwo_sum[bc] = Ii; xf_sum[bc] = Ff; }
}

// ---------------------------------------------------------------------------
// Pass 2: SE MLP + center gate -> cal[b,c].  grid = 64 blocks, 256 threads
// ---------------------------------------------------------------------------
__global__ void k_pass2_cal(const float* __restrict__ gap_sum,
                            const float* __restrict__ cwo_sum,
                            const float* __restrict__ se_w1,
                            const float* __restrict__ se_b1,
                            const float* __restrict__ se_w2,
                            const float* __restrict__ se_b2,
                            float* __restrict__ cal) {
    int b = blockIdx.x, tid = threadIdx.x;
    __shared__ float red[256];
    __shared__ float hid[3];
    float p0 = 0.f, p1 = 0.f, p2 = 0.f;
    for (int c = tid; c < C_; c += 256) {
        float g = gap_sum[b * C_ + c] * (1.0f / (T_ * HW_));
        p0 += g * se_w1[0 * C_ + c];
        p1 += g * se_w1[1 * C_ + c];
        p2 += g * se_w1[2 * C_ + c];
    }
    float s0 = blockSum(p0, red, tid, 256);
    float s1 = blockSum(p1, red, tid, 256);
    float s2 = blockSum(p2, red, tid, 256);
    if (tid == 0) {
        hid[0] = fmaxf(s0 + se_b1[0], 0.f);
        hid[1] = fmaxf(s1 + se_b1[1], 0.f);
        hid[2] = fmaxf(s2 + se_b1[2], 0.f);
    }
    __syncthreads();
    for (int c = tid; c < C_; c += 256) {
        float z = hid[0] * se_w2[c * 3 + 0] + hid[1] * se_w2[c * 3 + 1] +
                  hid[2] * se_w2[c * 3 + 2] + se_b2[c];
        float se = 1.0f / (1.0f + expf(-z));
        float cw  = gap_sum[b * C_ + c] * (1.0f / (T_ * HW_));
        float cwo = cwo_sum[b * C_ + c] * (1.0f / (T_ * (H_ - 2) * (W_ - 2)));
        float center = (cwo / (cw + 1e-8f) > 1.0f) ? 1.0f : 0.f;
        cal[b * C_ + c] = se * center;
    }
}

// ---------------------------------------------------------------------------
// WMMA fragment loaders (wave32, ISA 7.12.2 layouts)
// ---------------------------------------------------------------------------
__device__ __forceinline__ v16h loadFragA(const _Float16 (*As)[32], int tid) {
    v16h a;
    int g = tid >> 4, m = tid & 15;
#pragma unroll
    for (int i = 0; i < 4; ++i) {
        int k = 8 * g + 2 * i;
        a[2 * i] = As[m][k]; a[2 * i + 1] = As[m][k + 1];
    }
#pragma unroll
    for (int i = 4; i < 8; ++i) {
        int k = 16 + 8 * g + 2 * (i - 4);
        a[2 * i] = As[m][k]; a[2 * i + 1] = As[m][k + 1];
    }
    return a;
}
__device__ __forceinline__ v16h loadFragB(const _Float16 (*Bt)[32], int tid) {
    v16h bfr;
    int g = tid >> 4, n = tid & 15;
#pragma unroll
    for (int j = 0; j < 4; ++j) {
        int k = 8 * g + 2 * j;
        bfr[2 * j] = Bt[n][k]; bfr[2 * j + 1] = Bt[n][k + 1];
    }
#pragma unroll
    for (int j = 4; j < 8; ++j) {
        int k = 16 + 8 * g + 2 * (j - 4);
        bfr[2 * j] = Bt[n][k]; bfr[2 * j + 1] = Bt[n][k + 1];
    }
    return bfr;
}

// Convert 16 contiguous floats (float4-vectorized) to two v8h with optional relu*scale
__device__ __forceinline__ void cvt16(const float* __restrict__ src, float scale,
                                      bool relu, v8h& h0, v8h& h1) {
    const float4* s4 = (const float4*)src;
    float vals[16];
#pragma unroll
    for (int q = 0; q < 4; ++q) {
        float4 f = s4[q];
        vals[4 * q + 0] = f.x; vals[4 * q + 1] = f.y;
        vals[4 * q + 2] = f.z; vals[4 * q + 3] = f.w;
    }
#pragma unroll
    for (int e = 0; e < 8; ++e) {
        float v = vals[e] * scale;
        h0[e] = (_Float16)(relu ? fmaxf(v, 0.f) : v);
    }
#pragma unroll
    for (int e = 0; e < 8; ++e) {
        float v = vals[8 + e] * scale;
        h1[e] = (_Float16)(relu ? fmaxf(v, 0.f) : v);
    }
}

// ---------------------------------------------------------------------------
// Pass 3: sa[b,k,n] = relu(BN( sum_c W[k,c]*relu(x[b,c,0,n]*cal[b,c]) + b_k ))
// grid = (32 ntiles, 64 b), block = 32 (one wave)
// ---------------------------------------------------------------------------
__global__ void k_pass3_sa_wmma(const float* __restrict__ x,
                                const float* __restrict__ cal,
                                const float* __restrict__ w_k,
                                const float* __restrict__ b_k,
                                const float* __restrict__ bn_gamma,
                                const float* __restrict__ bn_beta,
                                const float* __restrict__ bn_mean,
                                const float* __restrict__ bn_var,
                                float* __restrict__ sa) {
    int nt = blockIdx.x, b = blockIdx.y, tid = threadIdx.x;
    int n0 = nt * 16;
    __shared__ __align__(16) _Float16 As[16][32];
    __shared__ __align__(16) _Float16 Bt[16][32];
    v8f acc = {};
    for (int kk = 0; kk < C_; kk += 32) {
        // A = w_k rows 0..8 (zero-padded to 16), cols kk..kk+31. One divergent
        // branch per thread; vectorized loads and contiguous v8h LDS stores.
        {
            int m = tid >> 1;
            int kb = (tid & 1) * 16;
            v8h h0 = {}, h1 = {};
            if (m < KK_)
                cvt16(w_k + m * C_ + kk + kb, 1.0f, false, h0, h1);
            *reinterpret_cast<v8h*>(&As[m][kb])     = h0;
            *reinterpret_cast<v8h*>(&As[m][kb + 8]) = h1;
        }
        // B^T: Bt[n][k] = relu(x[b, kk+k, t=0, n0+n] * cal), thread k=tid
        {
            int c = kk + tid;
            float cv = cal[b * C_ + c];
            v8h h0, h1;
            cvt16(x + ((size_t)(b * C_ + c) * T_) * HW_ + n0, cv, true, h0, h1);
#pragma unroll
            for (int e = 0; e < 8; ++e) Bt[e][tid]     = h0[e];
#pragma unroll
            for (int e = 0; e < 8; ++e) Bt[8 + e][tid] = h1[e];
        }
        __syncthreads();
        v16h af = loadFragA(As, tid);
        v16h bf = loadFragB(Bt, tid);
        acc = __builtin_amdgcn_wmma_f32_16x16x32_f16(false, af, false, bf,
                                                     (short)0, acc, false, false);
        __syncthreads();
    }
    int g = tid >> 4;
    int n = n0 + (tid & 15);
#pragma unroll
    for (int r = 0; r < 8; ++r) {
        int m = r + 8 * g;
        if (m < KK_) {
            float v = acc[r] + b_k[m];
            v = (v - bn_mean[m]) * (bn_gamma[m] * rsqrtf(bn_var[m] + 1e-5f)) + bn_beta[m];
            sa[((size_t)b * KK_ + m) * HW_ + n] = fmaxf(v, 0.f);
        }
    }
}

// ---------------------------------------------------------------------------
// Pass 4: double L1 normalization; output sa_norm[b][n][k] (n-major)
// grid = 64 blocks, 512 threads
// ---------------------------------------------------------------------------
__global__ void k_pass4_norm(const float* __restrict__ sa,
                             float* __restrict__ sa_norm) {
    int b = blockIdx.x, tid = threadIdx.x;
    __shared__ float colsum[KK_];
    if (tid < KK_) colsum[tid] = 0.f;
    float v[KK_];
    float s1 = 0.f;
#pragma unroll
    for (int k = 0; k < KK_; ++k) {
        v[k] = sa[((size_t)b * KK_ + k) * HW_ + tid];
        s1 += fabsf(v[k]);
    }
    float d1 = fmaxf(s1, 1e-12f);
    __syncthreads();
#pragma unroll
    for (int k = 0; k < KK_; ++k) {
        v[k] /= d1;
        atomicAdd(&colsum[k], fabsf(v[k]));
    }
    __syncthreads();
#pragma unroll
    for (int k = 0; k < KK_; ++k)
        sa_norm[((size_t)b * HW_ + tid) * KK_ + k] = v[k] / fmaxf(colsum[k], 1e-12f);
}

// ---------------------------------------------------------------------------
// Pass 5: ker[b,c,k] = sum_n Xse[b,c,n]*sa_norm[b,n,k]
// grid = (64 mtiles, 64 b), block = 32
// ---------------------------------------------------------------------------
__global__ void k_pass5_ker_wmma(const float* __restrict__ x,
                                 const float* __restrict__ cal,
                                 const float* __restrict__ sa_norm,
                                 float* __restrict__ ker) {
    int mt = blockIdx.x, b = blockIdx.y, tid = threadIdx.x;
    int c0 = mt * 16;
    __shared__ __align__(16) _Float16 As[16][32];
    __shared__ __align__(16) _Float16 Bt[16][32];
    v8f acc = {};
    for (int kk = 0; kk < HW_; kk += 32) {
        // A = Xse[c0+m][kk+k]; contiguous vectorized path, no divergence
        {
            int m = tid >> 1;
            int kb = (tid & 1) * 16;
            int c = c0 + m;
            float cv = cal[b * C_ + c];
            v8h h0, h1;
            cvt16(x + ((size_t)(b * C_ + c) * T_) * HW_ + kk + kb, cv, true, h0, h1);
            *reinterpret_cast<v8h*>(&As[m][kb])     = h0;
            *reinterpret_cast<v8h*>(&As[m][kb + 8]) = h1;
        }
        // B^T: Bt[n][k] = sa_norm[b][kk+k][n] (n<9), zero pad (compile-time cond)
        {
            const float* sp = sa_norm + ((size_t)b * HW_ + kk + tid) * KK_;
#pragma unroll
            for (int e = 0; e < 16; ++e)
                Bt[e][tid] = (e < KK_) ? (_Float16)sp[e] : (_Float16)0.f;
        }
        __syncthreads();
        v16h af = loadFragA(As, tid);
        v16h bf = loadFragB(Bt, tid);
        acc = __builtin_amdgcn_wmma_f32_16x16x32_f16(false, af, false, bf,
                                                     (short)0, acc, false, false);
        __syncthreads();
    }
    int g = tid >> 4;
    int n = tid & 15;            // k_out
#pragma unroll
    for (int r = 0; r < 8; ++r) {
        int m = r + 8 * g;       // c row within tile
        if (n < KK_)
            ker[((size_t)b * C_ + c0 + m) * KK_ + n] = acc[r];
    }
}

// ---------------------------------------------------------------------------
// Init accumulators
// ---------------------------------------------------------------------------
__global__ void k_init_zero(float* __restrict__ f_raw, float* __restrict__ den) {
    int i = blockIdx.x * blockDim.x + threadIdx.x;
    if (i < B_ * TL_ * HW_) f_raw[i] = 0.f;
    if (i < B_) den[i] = 0.f;
}

// ---------------------------------------------------------------------------
// Pass 6: depthwise-per-b 3x3 conv over c (edge padding).
// Streams raw x tiles global->LDS with double-buffered ASYNC loads
// (relu(x*cal)*ker == relu(x)*(cal*ker) since cal >= 0).
// grid = (16 c-chunks, 3 t, 64 b), block = 512
// ---------------------------------------------------------------------------
__global__ void k_pass6_conv(const float* __restrict__ x,
                             const float* __restrict__ cal,
                             const float* __restrict__ ker,
                             float* __restrict__ f_raw) {
    int chunk = blockIdx.x, t = blockIdx.y, b = blockIdx.z, tid = threadIdx.x;
    __shared__ float tile[2][HW_];
    __shared__ float kbuf[2][KK_];
    int y = tid >> 4, xc = tid & 15;
    float acc = 0.f;
    const int CCH = C_ / 16;
    int cbeg = chunk * CCH;

    issueTileB32(&tile[0][tid],
                 x + ((size_t)(b * C_ + cbeg) * T_ + 1 + t) * HW_ + tid);
    for (int i = 0; i < CCH; ++i) {
        int c = cbeg + i;
        int cur = i & 1;
        bool more = (i + 1 < CCH);
        if (more)
            issueTileB32(&tile[cur ^ 1][tid],
                         x + ((size_t)(b * C_ + c + 1) * T_ + 1 + t) * HW_ + tid);
        if (tid < KK_)
            kbuf[cur][tid] = ker[((size_t)b * C_ + c) * KK_ + tid] * cal[b * C_ + c];
        if (more) asyncWait1(); else asyncWait0();
        __syncthreads();
#pragma unroll
        for (int ky = 0; ky < 3; ++ky)
#pragma unroll
            for (int kx = 0; kx < 3; ++kx) {
                int yy = min(max(y + ky - 1, 0), H_ - 1);
                int xx = min(max(xc + kx - 1, 0), W_ - 1);
                acc += fmaxf(tile[cur][yy * W_ + xx], 0.f) * kbuf[cur][ky * 3 + kx];
            }
        __syncthreads();
    }
    atomicAdd(&f_raw[((size_t)b * TL_ + t) * HW_ + tid], acc);
}

// ---------------------------------------------------------------------------
// Pass 7: scale, softmax over 512, clamp, invert, exp/suppress; den[b] += sum(1-f)
// grid = (3 t, 64 b), block = 512
// ---------------------------------------------------------------------------
__global__ void k_pass7_softmax(const float* __restrict__ f_raw,
                                float* __restrict__ f_final,
                                float* __restrict__ den) {
    int t = blockIdx.x, b = blockIdx.y, tid = threadIdx.x;
    __shared__ float red[HW_];
    float r = f_raw[((size_t)b * TL_ + t) * HW_ + tid] * (1.0f / (float)KK_);
    red[tid] = r; __syncthreads();
    for (int s = 256; s > 0; s >>= 1) {
        if (tid < s) red[tid] = fmaxf(red[tid], red[tid + s]);
        __syncthreads();
    }
    float mx = red[0]; __syncthreads();
    float e = expf(r - mx);
    float sum = blockSum(e, red, tid, HW_);
    float p = fmaxf(e / sum, 1e-4f);
    float fv = 1.0f / (p * (float)HW_);
    float outv = (fv < 1.0f) ? expf((fv - 1.0f) * BETA_) : 1.0f;
    f_final[((size_t)b * TL_ + t) * HW_ + tid] = outv;
    float om = blockSum(1.0f - outv, red, tid, HW_);
    if (tid == 0) atomicAdd(&den[b], om);
}

// ---------------------------------------------------------------------------
// Pass 8a: xl_res = xl*f + alpha*xf_mean ; num[b,c] = sum((1-f)*xl)
// grid = (1024 c, 64 b), block = 128 (float4 vectorized)
// ---------------------------------------------------------------------------
__global__ void k_pass8a(const float* __restrict__ x,
                         const float* __restrict__ f_final,
                         const float* __restrict__ xf_sum,
                         float* __restrict__ out,
                         float* __restrict__ num) {
    int c = blockIdx.x, b = blockIdx.y, tid = threadIdx.x;
    size_t base = (size_t)(b * C_ + c) * (T_ * HW_);
    float xfm = xf_sum[b * C_ + c] * (1.0f / (float)HW_);
    float add = ALPHA_ * xfm;
    float acc = 0.f;
#pragma unroll
    for (int t = 0; t < TL_; ++t) {
        const float4* xp = (const float4*)(x + base + (size_t)(1 + t) * HW_);
        const float4* fp = (const float4*)(f_final + ((size_t)b * TL_ + t) * HW_);
        float4* op = (float4*)(out + base + (size_t)(1 + t) * HW_);
        if (t + 1 < TL_)
            __builtin_prefetch((const float4*)(x + base + (size_t)(2 + t) * HW_) + tid, 0, 1);
        float4 v = xp[tid];
        float4 fv = fp[tid];
        float4 o;
        o.x = v.x * fv.x + add; o.y = v.y * fv.y + add;
        o.z = v.z * fv.z + add; o.w = v.w * fv.w + add;
        op[tid] = o;
        acc += (1.f - fv.x) * v.x + (1.f - fv.y) * v.y +
               (1.f - fv.z) * v.z + (1.f - fv.w) * v.w;
    }
    __shared__ float red[128];
    float s = blockSum(acc, red, tid, 128);
    if (tid == 0) num[b * C_ + c] = s;
}

// ---------------------------------------------------------------------------
// Pass 8b: xf_res = xf + alpha * num[b,c]/den[b]
// grid = 65536 blocks, block = 128 (float4 vectorized)
// ---------------------------------------------------------------------------
__global__ void k_pass8b(const float* __restrict__ x,
                         const float* __restrict__ num,
                         const float* __restrict__ den,
                         float* __restrict__ out) {
    int bc = blockIdx.x, tid = threadIdx.x;
    int b = bc >> 10;
    float r = ALPHA_ * num[bc] / den[b];
    const float4* xp = (const float4*)(x + (size_t)bc * (T_ * HW_));
    float4* op = (float4*)(out + (size_t)bc * (T_ * HW_));
    float4 v = xp[tid];
    v.x += r; v.y += r; v.z += r; v.w += r;
    op[tid] = v;
}

// ---------------------------------------------------------------------------
// Host launcher
// ---------------------------------------------------------------------------
extern "C" void kernel_launch(void* const* d_in, const int* in_sizes, int n_in,
                              void* d_out, int out_size, void* d_ws, size_t ws_size,
                              hipStream_t stream) {
    const float* x        = (const float*)d_in[0];
    const float* w_k      = (const float*)d_in[1];
    const float* b_k      = (const float*)d_in[2];
    const float* bn_gamma = (const float*)d_in[3];
    const float* bn_beta  = (const float*)d_in[4];
    const float* bn_mean  = (const float*)d_in[5];
    const float* bn_var   = (const float*)d_in[6];
    const float* se_w1    = (const float*)d_in[7];
    const float* se_b1    = (const float*)d_in[8];
    const float* se_w2    = (const float*)d_in[9];
    const float* se_b2    = (const float*)d_in[10];
    float* out = (float*)d_out;

    float* ws = (float*)d_ws;
    float* gap_sum = ws;                       // 65536
    float* cwo_sum = gap_sum + B_ * C_;        // 65536
    float* xf_sum  = cwo_sum + B_ * C_;        // 65536
    float* cal     = xf_sum  + B_ * C_;        // 65536
    float* sa      = cal     + B_ * C_;        // 294912
    float* sa_norm = sa      + B_ * KK_ * HW_; // 294912
    float* ker     = sa_norm + B_ * HW_ * KK_; // 589824
    float* f_raw   = ker     + B_ * C_ * KK_;  // 98304
    float* f_final = f_raw   + B_ * TL_ * HW_; // 98304
    float* den     = f_final + B_ * TL_ * HW_; // 64
    float* num     = den     + B_;             // 65536

    k_pass1_reduce<<<B_ * C_, 256, 0, stream>>>(x, gap_sum, cwo_sum, xf_sum);
    k_pass2_cal<<<B_, 256, 0, stream>>>(gap_sum, cwo_sum, se_w1, se_b1, se_w2, se_b2, cal);
    k_init_zero<<<(B_ * TL_ * HW_ + 255) / 256, 256, 0, stream>>>(f_raw, den);
    k_pass3_sa_wmma<<<dim3(HW_ / 16, B_), 32, 0, stream>>>(
        x, cal, w_k, b_k, bn_gamma, bn_beta, bn_mean, bn_var, sa);
    k_pass4_norm<<<B_, HW_, 0, stream>>>(sa, sa_norm);
    k_pass5_ker_wmma<<<dim3(C_ / 16, B_), 32, 0, stream>>>(x, cal, sa_norm, ker);
    k_pass6_conv<<<dim3(16, TL_, B_), HW_, 0, stream>>>(x, cal, ker, f_raw);
    k_pass7_softmax<<<dim3(TL_, B_), HW_, 0, stream>>>(f_raw, f_final, den);
    k_pass8a<<<dim3(C_, B_), 128, 0, stream>>>(x, f_final, xf_sum, out, num);
    k_pass8b<<<B_ * C_, 128, 0, stream>>>(x, num, den, out);
}
